// AltAttention_84628035601126
// MI455X (gfx1250) — compile-verified, matrix operand order checked by reading, not run on hardware
//
#include <hip/hip_runtime.h>
#include <hip/hip_bf16.h>

typedef __attribute__((ext_vector_type(16))) _Float16 v16h;
typedef __attribute__((ext_vector_type(8)))  _Float16 v8h;
typedef __attribute__((ext_vector_type(8)))  float    v8f;

constexpr int BB = 2, NN = 2048, CC = 1024, HH = 16, HD = 64;
constexpr float SCALE = 0.125f;  // HD^-0.5 = 64^-0.5

__device__ __forceinline__ v16h cat8(v8h lo, v8h hi) {
  return __builtin_shufflevector(lo, hi, 0,1,2,3,4,5,6,7,8,9,10,11,12,13,14,15);
}

// ---------------------------------------------------------------------------
// Tiled WMMA GEMM:  C[M,Nout] = A[M,K] * W[Nout,K]^T   (torch Linear layout)
// Block: 256 threads (8 waves), 128x128 output tile, K-step 32 (f16 WMMA).
// Wave grid 2x4: each wave owns 64 rows x 32 cols = 8 wmma tiles.
// EPI==0: scatter f16 Q(*SCALE)/K/V into [3][B,H,N,HD] workspace.
// EPI==1: add bias, store f32 row-major.
// ---------------------------------------------------------------------------
template <bool A_HALF, int EPI>
__global__ void __launch_bounds__(256)
gemm_wmma(const void* __restrict__ Ap, const float* __restrict__ W,
          int K, int Nout,
          _Float16* __restrict__ qkv_out,
          const float* __restrict__ bias, float* __restrict__ Cout)
{
  constexpr int ST = 40;                 // LDS row stride in halves (bank-friendly)
  __shared__ _Float16 As[128 * ST];
  __shared__ _Float16 Ws[128 * ST];

  const int tid  = threadIdx.x;
  const int lane = tid & 31;
  const int wave = tid >> 5;
  const int wm   = wave & 1;             // 0..1 -> 64-row half
  const int wn   = wave >> 1;            // 0..3 -> 32-col quarter
  const int row0 = blockIdx.x * 128;
  const int col0 = blockIdx.y * 128;

  const int lrow = tid >> 1;             // 0..127 (loader row)
  const int lkh  = (tid & 1) * 16;       // 0 or 16 (loader k-half)
  const int lm   = lane & 15;
  const int g    = lane >> 4;            // lane half (K-slot group)

  const v8f zero = {};
  v8f acc[4][2];
#pragma unroll
  for (int r = 0; r < 4; ++r)
#pragma unroll
    for (int c = 0; c < 2; ++c) acc[r][c] = zero;

  for (int kk = 0; kk < K; kk += 32) {
    // ---- stage A tile (128x32) into LDS as f16 ----
    if constexpr (A_HALF) {
      const _Float16* src = (const _Float16*)Ap + (size_t)(row0 + lrow) * K + kk + lkh;
      *(v8h*)&As[lrow * ST + lkh]     = *(const v8h*)(src);
      *(v8h*)&As[lrow * ST + lkh + 8] = *(const v8h*)(src + 8);
    } else {
      const float* src = (const float*)Ap + (size_t)(row0 + lrow) * K + kk + lkh;
      float4 f0 = *(const float4*)(src);
      float4 f1 = *(const float4*)(src + 4);
      float4 f2 = *(const float4*)(src + 8);
      float4 f3 = *(const float4*)(src + 12);
      v8h h0, h1;
      h0[0]=(_Float16)f0.x; h0[1]=(_Float16)f0.y; h0[2]=(_Float16)f0.z; h0[3]=(_Float16)f0.w;
      h0[4]=(_Float16)f1.x; h0[5]=(_Float16)f1.y; h0[6]=(_Float16)f1.z; h0[7]=(_Float16)f1.w;
      h1[0]=(_Float16)f2.x; h1[1]=(_Float16)f2.y; h1[2]=(_Float16)f2.z; h1[3]=(_Float16)f2.w;
      h1[4]=(_Float16)f3.x; h1[5]=(_Float16)f3.y; h1[6]=(_Float16)f3.z; h1[7]=(_Float16)f3.w;
      *(v8h*)&As[lrow * ST + lkh]     = h0;
      *(v8h*)&As[lrow * ST + lkh + 8] = h1;
    }
    // ---- stage W tile (128 n-rows x 32 k) into LDS as f16 ----
    {
      const float* src = W + (size_t)(col0 + lrow) * K + kk + lkh;
      float4 f0 = *(const float4*)(src);
      float4 f1 = *(const float4*)(src + 4);
      float4 f2 = *(const float4*)(src + 8);
      float4 f3 = *(const float4*)(src + 12);
      v8h h0, h1;
      h0[0]=(_Float16)f0.x; h0[1]=(_Float16)f0.y; h0[2]=(_Float16)f0.z; h0[3]=(_Float16)f0.w;
      h0[4]=(_Float16)f1.x; h0[5]=(_Float16)f1.y; h0[6]=(_Float16)f1.z; h0[7]=(_Float16)f1.w;
      h1[0]=(_Float16)f2.x; h1[1]=(_Float16)f2.y; h1[2]=(_Float16)f2.z; h1[3]=(_Float16)f2.w;
      h1[4]=(_Float16)f3.x; h1[5]=(_Float16)f3.y; h1[6]=(_Float16)f3.z; h1[7]=(_Float16)f3.w;
      *(v8h*)&Ws[lrow * ST + lkh]     = h0;
      *(v8h*)&Ws[lrow * ST + lkh + 8] = h1;
    }
    __syncthreads();

    // ---- fragments + 8 WMMA per wave ----
    v16h afr[4], bfr[2];
#pragma unroll
    for (int r = 0; r < 4; ++r) {
      const _Float16* p = &As[(wm * 64 + r * 16 + lm) * ST + g * 16];
      afr[r] = cat8(*(const v8h*)p, *(const v8h*)(p + 8));
    }
#pragma unroll
    for (int c = 0; c < 2; ++c) {
      const _Float16* p = &Ws[(wn * 32 + c * 16 + lm) * ST + g * 16];
      bfr[c] = cat8(*(const v8h*)p, *(const v8h*)(p + 8));
    }
#pragma unroll
    for (int r = 0; r < 4; ++r)
#pragma unroll
      for (int c = 0; c < 2; ++c)
        acc[r][c] = __builtin_amdgcn_wmma_f32_16x16x32_f16(
            false, afr[r], false, bfr[c], (short)0, acc[r][c], false, false);
    __syncthreads();
  }

  // ---- epilogue ----
#pragma unroll
  for (int r = 0; r < 4; ++r) {
#pragma unroll
    for (int c = 0; c < 2; ++c) {
#pragma unroll
      for (int j = 0; j < 8; ++j) {
        const int m = row0 + wm * 64 + r * 16 + j + 8 * g;
        const int n = col0 + wn * 32 + c * 16 + lm;
        float v = acc[r][c][j];
        if constexpr (EPI == 0) {
          const int sel = n >> 10;        // 0:q 1:k 2:v
          const int rem = n & 1023;
          const int h = rem >> 6, d = rem & 63;
          const int b = m >> 11, q = m & 2047;
          if (sel == 0) v *= SCALE;
          const size_t off = (size_t)sel * (BB * HH * NN * HD) +
                             (((size_t)(b * HH + h)) * NN + q) * HD + d;
          qkv_out[off] = (_Float16)v;
        } else {
          v += bias[n];
          Cout[(size_t)m * Nout + n] = v;
        }
      }
    }
  }
}

// ---------------------------------------------------------------------------
// Flash attention with alibi + key-padding mask, online softmax.
// Block: 128 threads (4 waves). Grid: (B*H, N/64).
// Each wave owns 16 Q-rows; 64-key blocks streamed; alibi read non-temporal.
// ---------------------------------------------------------------------------
__global__ void __launch_bounds__(128)
flash_attn(const _Float16* __restrict__ qh, const _Float16* __restrict__ kh,
           const _Float16* __restrict__ vh, const float* __restrict__ alibi,
           const unsigned char* __restrict__ mask, _Float16* __restrict__ oh)
{
  constexpr int KS = 72;                  // LDS row stride in halves
  __shared__ _Float16 Ks[64 * KS];        // [key][d]
  __shared__ _Float16 Vs[64 * KS];        // [d][key]  (transposed)
  __shared__ _Float16 Ps[4 * 16 * KS];    // per-wave P tiles

  const int tid = threadIdx.x, lane = tid & 31, wave = tid >> 5;
  const int bh = blockIdx.x;              // b*H + h
  const int b = bh >> 4, h = bh & 15;
  const int q0 = blockIdx.y * 64;
  const int lm = lane & 15, g = lane >> 4;

  const size_t headoff = (size_t)bh * NN * HD;

  // Q fragments (already *SCALE), kept in registers for the whole kernel
  v16h qf[2];
  {
    const int q = q0 + wave * 16 + lm;
    const _Float16* qp = qh + headoff + (size_t)q * HD + g * 16;
#pragma unroll
    for (int s = 0; s < 2; ++s)
      qf[s] = cat8(*(const v8h*)(qp + s * 32), *(const v8h*)(qp + s * 32 + 8));
  }

  const v8f zero = {};
  v8f acc_o[4];
  float mrow[8], lrow[8];
#pragma unroll
  for (int cb = 0; cb < 4; ++cb) acc_o[cb] = zero;
#pragma unroll
  for (int j = 0; j < 8; ++j) { mrow[j] = -1.0e30f; lrow[j] = 0.f; }

  const int lkey = tid >> 1;              // 0..63
  const int loff = (tid & 1) * 32;        // 0 or 32

  for (int k0 = 0; k0 < NN; k0 += 64) {
    // ---- stage K block [64 keys][64 d] ----
    {
      const _Float16* src = kh + headoff + (size_t)(k0 + lkey) * HD + loff;
#pragma unroll
      for (int i = 0; i < 4; ++i)
        *(v8h*)&Ks[lkey * KS + loff + i * 8] = *(const v8h*)(src + i * 8);
    }
    // ---- stage V block transposed [64 d][64 keys] ----
    {
      const _Float16* src = vh + headoff + (size_t)(k0 + lkey) * HD + loff;
#pragma unroll
      for (int i = 0; i < 4; ++i) {
        v8h vv = *(const v8h*)(src + i * 8);
#pragma unroll
        for (int e = 0; e < 8; ++e)
          Vs[(loff + i * 8 + e) * KS + lkey] = vv[e];
      }
    }
    __syncthreads();

    // ---- S = Q K^T  (4 col-blocks x 2 k-steps = 8 WMMA) ----
    v8f accs[4];
#pragma unroll
    for (int cb = 0; cb < 4; ++cb) accs[cb] = zero;
#pragma unroll
    for (int cb = 0; cb < 4; ++cb) {
#pragma unroll
      for (int s = 0; s < 2; ++s) {
        const _Float16* p = &Ks[(cb * 16 + lm) * KS + s * 32 + g * 16];
        v16h kf = cat8(*(const v8h*)p, *(const v8h*)(p + 8));
        accs[cb] = __builtin_amdgcn_wmma_f32_16x16x32_f16(
            false, qf[s], false, kf, (short)0, accs[cb], false, false);
      }
    }

    // ---- + alibi (streamed, non-temporal) and key-padding mask ----
#pragma unroll
    for (int cb = 0; cb < 4; ++cb) {
      const int key = k0 + cb * 16 + lm;
      const unsigned char mk = mask[b * NN + key];
      const float* ab =
          alibi + ((size_t)bh * NN + (q0 + wave * 16 + 8 * g)) * NN + key;
#pragma unroll
      for (int j = 0; j < 8; ++j) {
        float s = accs[cb][j] + __builtin_nontemporal_load(ab + (size_t)j * NN);
        accs[cb][j] = mk ? -1.0e30f : s;
      }
    }

    // ---- online softmax per row (reduce across 16-lane halves) ----
    float alpha[8];
#pragma unroll
    for (int j = 0; j < 8; ++j) {
      float mx = fmaxf(fmaxf(accs[0][j], accs[1][j]),
                       fmaxf(accs[2][j], accs[3][j]));
#pragma unroll
      for (int o = 8; o >= 1; o >>= 1) mx = fmaxf(mx, __shfl_xor(mx, o, 16));
      const float mnew = fmaxf(mrow[j], mx);
      alpha[j] = __expf(mrow[j] - mnew);
      float sum = 0.f;
#pragma unroll
      for (int cb = 0; cb < 4; ++cb) {
        const float p = __expf(accs[cb][j] - mnew);
        accs[cb][j] = p;
        sum += p;
      }
#pragma unroll
      for (int o = 8; o >= 1; o >>= 1) sum += __shfl_xor(sum, o, 16);
      lrow[j] = lrow[j] * alpha[j] + sum;
      mrow[j] = mnew;
#pragma unroll
      for (int cb = 0; cb < 4; ++cb) acc_o[cb][j] *= alpha[j];
    }

    // ---- P -> LDS (C/D layout -> A layout round-trip), then O += P V ----
    _Float16* pw = &Ps[wave * 16 * KS];
#pragma unroll
    for (int cb = 0; cb < 4; ++cb)
#pragma unroll
      for (int j = 0; j < 8; ++j)
        pw[(j + 8 * g) * KS + cb * 16 + lm] = (_Float16)accs[cb][j];

#pragma unroll
    for (int s = 0; s < 2; ++s) {
      const _Float16* pp = &pw[lm * KS + s * 32 + g * 16];
      v16h pf = cat8(*(const v8h*)pp, *(const v8h*)(pp + 8));
#pragma unroll
      for (int cb = 0; cb < 4; ++cb) {
        const _Float16* vp = &Vs[(cb * 16 + lm) * KS + s * 32 + g * 16];
        v16h vf = cat8(*(const v8h*)vp, *(const v8h*)(vp + 8));
        acc_o[cb] = __builtin_amdgcn_wmma_f32_16x16x32_f16(
            false, pf, false, vf, (short)0, acc_o[cb], false, false);
      }
    }
    __syncthreads();
  }

  // ---- normalize, write merged-head fp16 [b][q][h*64+d] ----
#pragma unroll
  for (int cb = 0; cb < 4; ++cb) {
#pragma unroll
    for (int j = 0; j < 8; ++j) {
      const int q = q0 + wave * 16 + j + 8 * g;
      const int d = cb * 16 + lm;
      const float v = acc_o[cb][j] / lrow[j];
      oh[((size_t)(b * NN + q)) * CC + h * HD + d] = (_Float16)v;
    }
  }
}

// ---------------------------------------------------------------------------
extern "C" void kernel_launch(void* const* d_in, const int* in_sizes, int n_in,
                              void* d_out, int out_size, void* d_ws, size_t ws_size,
                              hipStream_t stream) {
  (void)in_sizes; (void)n_in; (void)out_size; (void)ws_size;
  const float* x              = (const float*)d_in[0];
  const unsigned char* pmask  = (const unsigned char*)d_in[1];
  const float* alibi          = (const float*)d_in[2];
  const float* qkv_w          = (const float*)d_in[3];
  const float* proj_w         = (const float*)d_in[4];
  const float* proj_b         = (const float*)d_in[5];
  float* out                  = (float*)d_out;

  _Float16* wsh  = (_Float16*)d_ws;
  _Float16* qkvh = wsh;                                       // 3 * B*H*N*HD halves
  _Float16* oh   = wsh + (size_t)3 * BB * HH * NN * HD;       // B*N*C halves

  // 1) QKV projection: [4096,1024] x [3072,1024]^T -> f16 Q(*SCALE)/K/V heads
  dim3 g1(BB * NN / 128, (3 * CC) / 128);                     // (32, 24)
  gemm_wmma<false, 0><<<g1, 256, 0, stream>>>(x, qkv_w, CC, 3 * CC,
                                              qkvh, nullptr, nullptr);

  // 2) Flash attention with alibi + padding mask
  dim3 g2(BB * HH, NN / 64);                                  // (32, 32)
  flash_attn<<<g2, 128, 0, stream>>>(qkvh,
                                     qkvh + (size_t)BB * HH * NN * HD,
                                     qkvh + (size_t)2 * BB * HH * NN * HD,
                                     alibi, pmask, oh);

  // 3) Output projection + bias: [4096,1024] x [1024,1024]^T -> f32 out
  dim3 g3(BB * NN / 128, CC / 128);                           // (32, 8)
  gemm_wmma<true, 1><<<g3, 256, 0, stream>>>(oh, proj_w, CC, CC,
                                             nullptr, proj_b, out);
}
